// LSTM_38714835206516
// MI455X (gfx1250) — 8-bit matrix operand order repaired and checked by reading, compiled, not run on hardware
//
#include <hip/hip_runtime.h>
#include <cstdint>


#define TSTEPS 784
#define HID    256
#define BATCH  512
#define NCLS   10
#define TILE_B 16
#define HAPITCH 264

typedef __attribute__((ext_vector_type(16))) int   v16i;
typedef __attribute__((ext_vector_type(8)))  float v8f;

__device__ __forceinline__ unsigned char f32_to_e4m3_sw(float f) {
  uint32_t u  = __float_as_uint(f);
  uint32_t s  = (u >> 24) & 0x80u;
  uint32_t au = u & 0x7fffffffu;
  if (au > 0x7f800000u) return (unsigned char)(s | 0x7fu);
  if (au >= 0x43e00000u) return (unsigned char)(s | 0x7eu);
  int e = (int)(au >> 23) - 127;
  if (e < -10) return (unsigned char)s;
  uint32_t mant = (au & 0x7fffffu) | 0x800000u;
  int shift = (e < -6) ? (14 - e) : 20;
  uint32_t keep  = mant >> shift;
  uint32_t rem   = mant & ((1u << shift) - 1u);
  uint32_t halfb = 1u << (shift - 1);
  keep += (rem > halfb || (rem == halfb && (keep & 1u))) ? 1u : 0u;
  uint32_t e8 = (e < -6) ? 0u : (uint32_t)(e + 7);
  if (e8 == 0u) {
    if (keep >= 8u) { e8 = 1u; keep -= 8u; }
  } else {
    if (keep >= 16u) { keep >>= 1; e8 += 1u; }
    keep &= 7u;
  }
  uint32_t val = (e8 << 3) | keep;
  if (val > 0x7eu) val = 0x7eu;
  return (unsigned char)(s | val);
}

__device__ __forceinline__ unsigned char f32_to_e4m3(float f) {
#if __has_builtin(__builtin_amdgcn_cvt_pk_fp8_f32)
  return (unsigned char)(__builtin_amdgcn_cvt_pk_fp8_f32(f, f, 0, false) & 0xff);
#else
  return f32_to_e4m3_sw(f);
#endif
}

__device__ __forceinline__ float fast_sigmoid(float x) {
  return __builtin_amdgcn_rcpf(1.f + __builtin_amdgcn_exp2f(x * -1.44269504088896f));
}
__device__ __forceinline__ float fast_tanh(float x) {
#if __has_builtin(__builtin_amdgcn_tanhf)
  return __builtin_amdgcn_tanhf(x);
#else
  return 2.f * __builtin_amdgcn_rcpf(1.f + __builtin_amdgcn_exp2f(x * -2.88539008177793f)) - 1.f;
#endif
}

__global__ __launch_bounds__(256) void pack_w(const float* __restrict__ Wg,
                                              const float* __restrict__ Wi,
                                              const float* __restrict__ Wf,
                                              const float* __restrict__ Wo,
                                              uint32_t* __restrict__ outp) {
  const int idx   = blockIdx.x * 256 + threadIdx.x;
  const int v     = idx & 7;
  const int lane  = (idx >> 3) & 31;
  const int frag  = idx >> 8;
  const int kf64  = frag & 3;
  const int ntile = frag >> 2;
  const int gate  = ntile >> 4;
  const int jcol  = ((ntile & 15) << 4) | (lane & 15);
  const int kbase = kf64 * 64 + ((v & 4) << 3) + (lane & 16) + ((v & 3) << 2);
  const float* W = (gate == 0) ? Wg : (gate == 1) ? Wi : (gate == 2) ? Wf : Wo;
  uint32_t d = 0;
#pragma unroll
  for (int b2 = 0; b2 < 4; ++b2)
    d |= ((uint32_t)f32_to_e4m3(W[(kbase + b2) * HID + jcol])) << (8 * b2);
  const int halfsel = kf64 >> 1;
  const int kfL     = kf64 & 1;
  outp[halfsel * 32768 + (ntile * 2 + kfL) * 256 + lane * 8 + v] = d;
}

__device__ __forceinline__ v16i load_bfrag(const uint32_t* p) {
  v16i r;
#pragma unroll
  for (int q = 0; q < 8; ++q) { r[q] = (int)p[q]; r[8 + q] = (int)p[256 + q]; }
  return r;
}

__global__ __launch_bounds__(256) void lstm_main(
    const float* __restrict__ x, const uint32_t* __restrict__ wpack,
    const float* __restrict__ Wgx, const float* __restrict__ bg,
    const float* __restrict__ Wix, const float* __restrict__ bi,
    const float* __restrict__ Wfx, const float* __restrict__ bfv,
    const float* __restrict__ Wox, const float* __restrict__ bo,
    const float* __restrict__ Wph, const float* __restrict__ bp,
    float* __restrict__ out) {
  __shared__ uint32_t wfragL[32768];
  __shared__ __align__(16) unsigned char hA[2][TILE_B * HAPITCH];
  __shared__ float hf32[TILE_B * HID];
  __shared__ float xs[2][TILE_B];

  const int tid  = threadIdx.x;
  const int lane = tid & 31;
  const int wid  = tid >> 5;
  const int row0 = blockIdx.x * TILE_B;
  const int col  = lane & 15;
  const int half = lane >> 4;
  const int wslice = wid * 32;

  {
    const uint4* src = (const uint4*)wpack;
    uint4* dst = (uint4*)wfragL;
    for (int i = tid; i < 32768 / 4; i += 256) dst[i] = src[i];
  }
  for (int i = tid; i < 2 * TILE_B * HAPITCH / 4; i += 256) ((uint32_t*)hA)[i] = 0u;
  if (tid < TILE_B) xs[0][tid] = x[(row0 + tid) * TSTEPS + 0];

  float xwr[2][4], bbr[2][4];
#pragma unroll
  for (int nt = 0; nt < 2; ++nt) {
    const int j = wslice + nt * 16 + col;
    xwr[nt][0] = Wgx[j]; xwr[nt][1] = Wix[j]; xwr[nt][2] = Wfx[j]; xwr[nt][3] = Wox[j];
    bbr[nt][0] = bg[j];  bbr[nt][1] = bi[j];  bbr[nt][2] = bfv[j]; bbr[nt][3] = bo[j];
  }
  __syncthreads();

  float creg[2][8] = {};
  const uint32_t* wpackG = wpack + 32768;

  for (int t = 0; t < TSTEPS; ++t) {
    const unsigned char* rd = hA[t & 1];
    unsigned char*       wr = hA[(t + 1) & 1];

    if (tid < TILE_B && (t + 1) < TSTEPS)
      xs[(t + 1) & 1][tid] = x[(row0 + tid) * TSTEPS + (t + 1)];

    const int m16  = col;
    const int kadd = half ? 16 : 0;
    v16i afr[2];
#pragma unroll
    for (int kf = 0; kf < 2; ++kf) {
#pragma unroll
      for (int v = 0; v < 8; ++v) {
        const int koff = ((v >> 2) << 5) + ((v & 3) << 2) + kadd;
        afr[kf][v]     = (int)*(const uint32_t*)&rd[m16 * HAPITCH + kf * 128 + koff];
        afr[kf][v + 8] = (int)*(const uint32_t*)&rd[m16 * HAPITCH + kf * 128 + 64 + koff];
      }
    }

    v8f acc[4][2] = {};
#pragma unroll
    for (int g = 0; g < 4; ++g) {
#pragma unroll
      for (int nt = 0; nt < 2; ++nt) {
        const int ntile = g * 16 + wid * 2 + nt;
        v16i b0 = load_bfrag(&wfragL[(ntile * 2) * 256 + lane * 8]);
        acc[g][nt] = __builtin_amdgcn_wmma_f32_16x16x128_fp8_fp8(
            afr[0], b0, (short)0, acc[g][nt], false, false);
        v16i b1 = load_bfrag(&wpackG[(ntile * 2) * 256 + lane * 8]);
        acc[g][nt] = __builtin_amdgcn_wmma_f32_16x16x128_fp8_fp8(
            afr[1], b1, (short)0, acc[g][nt], false, false);
      }
    }

    const float* xsr = xs[t & 1];
#pragma unroll
    for (int nt = 0; nt < 2; ++nt) {
      const int j = wslice + nt * 16 + col;
#pragma unroll
      for (int r = 0; r < 8; ++r) {
        const int m = r + half * 8;
        const float xv = xsr[m];
        const float pg = acc[0][nt][r] + xv * xwr[nt][0] + bbr[nt][0];
        const float pi = acc[1][nt][r] + xv * xwr[nt][1] + bbr[nt][1];
        const float pf = acc[2][nt][r] + xv * xwr[nt][2] + bbr[nt][2];
        const float po = acc[3][nt][r] + xv * xwr[nt][3] + bbr[nt][3];
        const float gg = fast_tanh(pg);
        const float ii = fast_sigmoid(pi);
        const float ff = fast_sigmoid(pf);
        const float oo = fast_sigmoid(po);
        const float c  = gg * ii + creg[nt][r] * ff;
        creg[nt][r] = c;
        const float h = fast_tanh(c) * oo;
        wr[m * HAPITCH + j] = f32_to_e4m3(h);
        if (t == TSTEPS - 1) hf32[m * HID + j] = h;
      }
    }
    __syncthreads();
  }

  if (tid < TILE_B * NCLS) {
    const int m = tid / NCLS, cc = tid % NCLS;
    float s = bp[cc];
    for (int j = 0; j < HID; ++j) s += hf32[m * HID + j] * Wph[j * NCLS + cc];
    out[(row0 + m) * NCLS + cc] = s;
  }
}

extern "C" void kernel_launch(void* const* d_in, const int* in_sizes, int n_in,
                              void* d_out, int out_size, void* d_ws, size_t ws_size,
                              hipStream_t stream) {
  (void)in_sizes; (void)n_in; (void)out_size; (void)ws_size;
  const float* x   = (const float*)d_in[0];
  const float* Wgx = (const float*)d_in[1];
  const float* Wgh = (const float*)d_in[2];
  const float* bg  = (const float*)d_in[3];
  const float* Wix = (const float*)d_in[4];
  const float* Wih = (const float*)d_in[5];
  const float* bi  = (const float*)d_in[6];
  const float* Wfx = (const float*)d_in[7];
  const float* Wfh = (const float*)d_in[8];
  const float* bfv = (const float*)d_in[9];
  const float* Wox = (const float*)d_in[10];
  const float* Woh = (const float*)d_in[11];
  const float* bo  = (const float*)d_in[12];
  const float* Wph = (const float*)d_in[13];
  const float* bp  = (const float*)d_in[14];
  uint32_t* wpack = (uint32_t*)d_ws;
  pack_w<<<256, 256, 0, stream>>>(Wgh, Wih, Wfh, Woh, wpack);
  lstm_main<<<BATCH / TILE_B, 256, 0, stream>>>(x, wpack, Wgx, bg, Wix, bi,
                                                Wfx, bfv, Wox, bo, Wph, bp,
                                                (float*)d_out);
}
